// SynapseNet_31147102830807
// MI455X (gfx1250) — matrix operand order repaired and checked by reading, not compiled, not run on hardware
//
#include <hip/hip_runtime.h>
#include <hip/hip_bf16.h>
#include <math.h>

// Problem constants (fixed by the reference)
#define LC   4
#define HC   8
#define DC   512
#define DFFC 2048
#define BC   2
#define SC   2048
#define DHC  64
#define MC   (BC*SC)     // 4096 rows of the token matrix

typedef __attribute__((ext_vector_type(16))) __bf16 bf16x16;
typedef __attribute__((ext_vector_type(8)))  float  f32x8;

#define WMMA_BF16(a,b,c) \
  __builtin_amdgcn_wmma_f32_16x16x32_bf16(false,(a),false,(b),(short)0,(c),false,false)

// ---- CDNA5 async global->LDS copy path (ASYNCcnt), with safe fallback -------
#if defined(__has_builtin)
# if __has_builtin(__builtin_amdgcn_global_load_async_to_lds_b128) && \
     __has_builtin(__builtin_amdgcn_s_wait_asynccnt)
#  define USE_ASYNC 1
# endif
#endif
#ifndef USE_ASYNC
# define USE_ASYNC 0
#endif

#if USE_ASYNC
// Builtin signature (from compiler diagnostic): param 0 is
// '__attribute__((__vector_size__(4*sizeof(int)))) int __device__ *' (AS1),
// param 1 is the LDS-side pointer (AS3), then two immediates (offset, cpol).
typedef int avec4i __attribute__((vector_size(16)));
static __device__ __forceinline__ void copy16_g2l(const unsigned short* g, unsigned short* l){
  __builtin_amdgcn_global_load_async_to_lds_b128(
      (__attribute__((address_space(1))) avec4i*)g,
      (__attribute__((address_space(3))) avec4i*)l, 0, 0);
}
static __device__ __forceinline__ void async_wait(){ __builtin_amdgcn_s_wait_asynccnt(0); }
#else
static __device__ __forceinline__ void copy16_g2l(const unsigned short* g, unsigned short* l){
  *(uint4*)l = *(const uint4*)g;
}
static __device__ __forceinline__ void async_wait(){}
#endif

// float -> bf16 bits, round to nearest even
static __device__ __forceinline__ unsigned short f2bf(float f){
  union{float f; unsigned u;} v; v.f=f;
  unsigned r = v.u + 0x7FFFu + ((v.u>>16)&1u);
  return (unsigned short)(r>>16);
}

// A-fragment (16x32 bf16) from LDS, row-major tile with 'stride' (in ushorts).
// CDNA5 layout: lanes 0-15 row M=lane, K 0..7 then 16..23; lanes 16-31 row M=lane-16, K 8..15 then 24..31.
static __device__ __forceinline__ bf16x16 fragA(const unsigned short* base,int row,int stride,int koff,int lane){
  const unsigned short* p = base + row*stride + koff + ((lane>>4)<<3);
  union{bf16x16 v; uint4 q[2];} u;
  u.q[0]=*(const uint4*)(p);
  u.q[1]=*(const uint4*)(p+16);
  return u.v;
}

// B-fragment (32x16 bf16) from LDS tile stored column-major: [n][k], 'stride' ushorts per column.
// CDNA5 layout: lanes 0-15 col N=lane, K 0..7 then 16..23; lanes 16-31 col N=lane-16, K 8..15 then 24..31.
static __device__ __forceinline__ bf16x16 fragB(const unsigned short* base,int col,int stride,int koff,int lane){
  const unsigned short* p = base + col*stride + koff + ((lane>>4)<<3);
  union{bf16x16 v; uint4 q[2];} u;
  u.q[0]=*(const uint4*)(p);
  u.q[1]=*(const uint4*)(p+16);
  return u.v;
}

// ------------------------------------------------- weight conversion (once per call)
// Row-major W[K][N] -> bf16, transposed to [N][K] (k-contiguous)
__global__ __launch_bounds__(256) void cvtT_kernel(const float* __restrict__ in,
    unsigned short* __restrict__ out, int K, int N){
  size_t idx = (size_t)blockIdx.x*256 + threadIdx.x;
  if(idx >= (size_t)K*N) return;
  int n = (int)(idx / K), k = (int)(idx % K);
  out[idx] = f2bf(in[(size_t)k*N + n]);
}
// Head-blocked W[H][K][64] -> bf16 [N][K] with n = h*64+e
__global__ __launch_bounds__(256) void cvtH_kernel(const float* __restrict__ in,
    unsigned short* __restrict__ out, int K){
  size_t idx = (size_t)blockIdx.x*256 + threadIdx.x;
  if(idx >= (size_t)K*DC) return;
  int n = (int)(idx / K), k = (int)(idx % K);
  int h = n>>6, e = n&63;
  out[idx] = f2bf(in[((size_t)h*K + k)*DHC + e]);
}

// ---------------------------------------------------------------- LayerNorm (bf16 out)
__global__ __launch_bounds__(256) void ln_kernel(const float* __restrict__ x,
    const float* __restrict__ w, const float* __restrict__ b, unsigned short* __restrict__ o){
  const int row = blockIdx.x;
  const float* xr = x + (size_t)row*DC;
  __shared__ float red[256];
  float s1=0.f, s2=0.f;
  for(int j=threadIdx.x; j<DC; j+=256){ float t=xr[j]; s1+=t; s2+=t*t; }
  red[threadIdx.x]=s1; __syncthreads();
  for(int off=128; off>0; off>>=1){ if(threadIdx.x<off) red[threadIdx.x]+=red[threadIdx.x+off]; __syncthreads(); }
  const float mean = red[0]*(1.f/DC);
  __syncthreads();
  red[threadIdx.x]=s2; __syncthreads();
  for(int off=128; off>0; off>>=1){ if(threadIdx.x<off) red[threadIdx.x]+=red[threadIdx.x+off]; __syncthreads(); }
  const float var = red[0]*(1.f/DC) - mean*mean;
  const float inv = rsqrtf(var + 1e-5f);
  for(int j=threadIdx.x; j<DC; j+=256)
    o[(size_t)row*DC + j] = f2bf((xr[j]-mean)*inv*w[j] + b[j]);
}

// ---------------------------------------------------------------- bf16-WMMA GEMM
// A: bf16 [M][K]; Bt: bf16 [N][K] (pre-transposed, k-contiguous)
// out = act(A x B + bias) + resid  -> Cb (bf16) if non-null, else C (f32)
// cmode: 0 = row-major [M,N]; 1 = head-major [b][h][s][e]  (n = h*64+e, m = b*S+s)
// act:   0 none, 1 SiLU, 2 exact GELU
__global__ __launch_bounds__(128) void gemm_wmma(const unsigned short* __restrict__ A,
    const unsigned short* __restrict__ Bt, float* __restrict__ C, unsigned short* __restrict__ Cb,
    int M, int N, int K, int cmode,
    const float* __restrict__ bias, const float* __restrict__ resid, int act){
  const int n0 = blockIdx.x*64, m0 = blockIdx.y*64;
  const int tid = threadIdx.x, lane = tid&31, w = tid>>5;
  __shared__ alignas(16) unsigned short As[2][64*32];   // [m][k]
  __shared__ alignas(16) unsigned short Bs[2][64*32];   // [n][k]
  f32x8 acc[4] = {};

  const int r0c = tid>>2,        c0c = (tid&3)<<3;       // thread's two uint4 slots
  const int r1c = (tid+128)>>2,  c1c = ((tid+128)&3)<<3;

  auto stage = [&](int k0, int buf){
    const unsigned short* ga0 = &A [(size_t)(m0+r0c)*K + k0 + c0c];
    const unsigned short* ga1 = &A [(size_t)(m0+r1c)*K + k0 + c1c];
    const unsigned short* gb0 = &Bt[(size_t)(n0+r0c)*K + k0 + c0c];
    const unsigned short* gb1 = &Bt[(size_t)(n0+r1c)*K + k0 + c1c];
    unsigned short* la0 = &As[buf][r0c*32+c0c];
    unsigned short* la1 = &As[buf][r1c*32+c1c];
    unsigned short* lb0 = &Bs[buf][r0c*32+c0c];
    unsigned short* lb1 = &Bs[buf][r1c*32+c1c];
#if USE_ASYNC
    copy16_g2l(ga0,la0); copy16_g2l(ga1,la1);
    copy16_g2l(gb0,lb0); copy16_g2l(gb1,lb1);
#else
    uint4 va0=*(const uint4*)ga0, va1=*(const uint4*)ga1;
    uint4 vb0=*(const uint4*)gb0, vb1=*(const uint4*)gb1;
    *(uint4*)la0=va0; *(uint4*)la1=va1;
    *(uint4*)lb0=vb0; *(uint4*)lb1=vb1;
    if(k0+32 < K){
      __builtin_prefetch(ga0 + 32, 0, 1);
      __builtin_prefetch(gb0 + 32, 0, 1);
    }
#endif
  };

  stage(0, 0);
  async_wait();
  __syncthreads();
  int buf = 0;
  for(int k0=0; k0<K; k0+=32, buf^=1){
    if(k0+32 < K) stage(k0+32, buf^1);       // overlap staging with WMMA
    bf16x16 bfr = fragB(Bs[buf], w*16+(lane&15), 32, 0, lane);
    #pragma unroll
    for(int mt=0; mt<4; ++mt){
      bf16x16 afr = fragA(As[buf], mt*16+(lane&15), 32, 0, lane);
      acc[mt] = WMMA_BF16(afr, bfr, acc[mt]);
    }
    async_wait();                            // next tile landed in LDS
    __syncthreads();
  }

  const int nn = n0 + w*16 + (lane&15);
  const int hi8 = (lane>>4)<<3;
  const float bsv = bias ? bias[nn] : 0.f;
  #pragma unroll
  for(int mt=0; mt<4; ++mt){
    #pragma unroll
    for(int r=0; r<8; ++r){
      int mm = m0 + mt*16 + hi8 + r;
      float v = acc[mt][r] + bsv;
      if(act==1)      v = v/(1.f+expf(-v));
      else if(act==2) v = 0.5f*v*(1.f+erff(v*0.70710678118f));
      if(resid) v += resid[(size_t)mm*N + nn];
      size_t off;
      if(cmode==0) off = (size_t)mm*N + nn;
      else{
        int b_=mm/SC, s_=mm%SC, h_=nn>>6, e_=nn&63;
        off = (((size_t)(b_*HC + h_))*SC + s_)*DHC + e_;
      }
      if(Cb) Cb[off] = f2bf(v);
      else   C [off] = v;
    }
  }
}

// ---------------------------------------------------------------- xPos rotary (f32 in -> packed bf16 out)
__global__ __launch_bounds__(256) void xpos_bf_kernel(const float* __restrict__ p,
    unsigned short* __restrict__ o, int down){
  const int idx = blockIdx.x*256 + threadIdx.x;       // pair index
  if(idx >= BC*HC*SC*(DHC/2)) return;
  const int j = idx & 31;
  const int s = (idx >> 5) % SC;
  const float* q2 = p + (size_t)idx*2;
  float base = (2.f*(float)j + 0.4f*(float)DHC)/(1.4f*(float)DHC);
  float sc   = exp2f(log2f(base) * ((float)s*(1.f/512.f)));
  if(down) sc = 1.f/sc;
  float invf = expf(-(float)j*(logf(10000.f)/32.f));
  float ang  = (float)s*invf;
  float sn = sinf(ang), cs = cosf(ang);
  float x1=q2[0], x2=q2[1];
  unsigned lo = f2bf((x1*cs - x2*sn)*sc);
  unsigned hi = f2bf((x2*cs + x1*sn)*sc);
  *(unsigned*)(o + (size_t)idx*2) = lo | (hi<<16);
}

// ---------------------------------------------------------------- causal retention (bf16 in)
// yh[b][h][s][e] = sum_t (q.k^T * gamma^(s-t) [s>=t]) v    — tiled 64x64, bf16 WMMA
__global__ __launch_bounds__(128) void attn_kernel(const unsigned short* __restrict__ qbf,
    const unsigned short* __restrict__ kbf, const unsigned short* __restrict__ vbf,
    float* __restrict__ yh){
  const int rt = blockIdx.x;                 // row tile (s)
  const int bh = blockIdx.y;                 // b*H+h
  const int h  = bh & (HC-1);
  const unsigned short* qt = qbf + (size_t)bh*SC*DHC + (size_t)rt*64*DHC; // contiguous 8KB tile
  const unsigned short* kp = kbf + (size_t)bh*SC*DHC;
  const unsigned short* vp = vbf + (size_t)bh*SC*DHC;
  float*                yp = yh  + (size_t)bh*SC*DHC;
  const int tid=threadIdx.x, lane=tid&31, w=tid>>5;
  __shared__ alignas(16) unsigned short Qs[64*64];     // [s][e]
  __shared__ alignas(16) unsigned short Ks[64*64];     // [t][e]
  __shared__ alignas(16) unsigned short Vt[64*64];     // [e][t] (transposed)
  __shared__ alignas(16) unsigned short atts[4][16*64];// per-wave masked scores [s][t]

  #pragma unroll
  for(int i=0;i<4;++i){                       // 512 x uint4, contiguous copy
    int idx = tid + i*128;
    copy16_g2l(qt + idx*8, &Qs[idx*8]);
  }
  const float l0 = logf(1.f/32.f), l1 = logf(1.f/512.f);
  const float gamma = 1.f - expf(l0 + (l1-l0)*((float)h/(float)(HC-1)));
  const float lg = log2f(gamma);
  const int hi8 = (lane>>4)<<3;

  f32x8 yacc[4] = {};
  for(int ct=0; ct<=rt; ++ct){
    __syncthreads();
    const unsigned short* kt = kp + (size_t)ct*64*DHC; // contiguous K tile
    #pragma unroll
    for(int i=0;i<4;++i){
      int idx = tid + i*128;
      copy16_g2l(kt + idx*8, &Ks[idx*8]);
    }
    const unsigned short* vt = vp + (size_t)ct*64*DHC;
    for(int idx=tid; idx<64*64; idx+=128){     // V staged transposed
      int r=idx>>6, c=idx&63;
      Vt[c*64+r] = vt[idx];
    }
    async_wait();
    __syncthreads();
    // scores: M=s(16 rows of this wave), N=t(64), K=e(64, two 32-chunks)
    f32x8 att[4] = {};
    for(int ec=0; ec<2; ++ec){
      bf16x16 a = fragA(Qs, w*16+(lane&15), 64, ec*32, lane);
      #pragma unroll
      for(int jt=0; jt<4; ++jt){
        bf16x16 b = fragB(Ks, jt*16+(lane&15), 64, ec*32, lane);
        att[jt] = WMMA_BF16(a, b, att[jt]);
      }
    }
    // apply causal decay mask on the accumulator (coords from C-layout), stash bf16
    const int sbase = rt*64 + w*16 + hi8;
    #pragma unroll
    for(int jt=0; jt<4; ++jt){
      #pragma unroll
      for(int r=0; r<8; ++r){
        int s_ = sbase + r;
        int t_ = ct*64 + jt*16 + (lane&15);
        int d_ = s_ - t_;
        float val = (d_>=0) ? att[jt][r]*exp2f(lg*(float)d_) : 0.f;
        atts[w][(hi8+r)*64 + jt*16 + (lane&15)] = f2bf(val);
      }
    }
    // yh += att x V : M=s(16), N=e(64), K=t(64, two 32-chunks); same-wave LDS reuse
    for(int tc=0; tc<2; ++tc){
      bf16x16 a2 = fragA(atts[w], (lane&15), 64, tc*32, lane);
      #pragma unroll
      for(int je=0; je<4; ++je){
        bf16x16 b2 = fragB(Vt, je*16+(lane&15), 64, tc*32, lane);
        yacc[je] = WMMA_BF16(a2, b2, yacc[je]);
      }
    }
  }
  #pragma unroll
  for(int je=0; je<4; ++je)
    #pragma unroll
    for(int r=0; r<8; ++r)
      yp[(size_t)(rt*64 + w*16 + hi8 + r)*DHC + je*16 + (lane&15)] = yacc[je][r];
}

// ---------------------------------------------------------------- per-head GroupNorm + SiLU gate (bf16 out)
__global__ __launch_bounds__(512) void gn_gate_kernel(const float* __restrict__ yh,
    const float* __restrict__ g, const float* __restrict__ gnw, const float* __restrict__ gnb,
    unsigned short* __restrict__ gy){
  const int m = blockIdx.x;                  // b*S+s
  const int b_ = m/SC, s_ = m%SC;
  const int n = threadIdx.x;                 // h*64+e
  const int h_ = n>>6, e_ = n&63;
  const float val = yh[(((size_t)(b_*HC + h_))*SC + s_)*DHC + e_];
  __shared__ float red[512];
  red[n]=val; __syncthreads();
  for(int off=32; off>0; off>>=1){ if(e_<off) red[n]+=red[n+off]; __syncthreads(); }
  const float mean = red[h_*64]*(1.f/DHC);
  __syncthreads();
  const float d = val-mean;
  red[n]=d*d; __syncthreads();
  for(int off=32; off>0; off>>=1){ if(e_<off) red[n]+=red[n+off]; __syncthreads(); }
  const float var = red[h_*64]*(1.f/DHC);
  const float yn = d*rsqrtf(var+1e-5f)*gnw[n] + gnb[n];
  gy[(size_t)m*DC + n] = f2bf(yn * g[(size_t)m*DC + n]);
}

// ---------------------------------------------------------------- host
static void launch_gemm(const unsigned short* A,const unsigned short* Bt,float* C,unsigned short* Cb,
                        int M,int N,int K,int cmode,const float* bias,const float* resid,int act,
                        hipStream_t st){
  dim3 grid(N/64, M/64);
  gemm_wmma<<<grid,128,0,st>>>(A,Bt,C,Cb,M,N,K,cmode,bias,resid,act);
}

extern "C" void kernel_launch(void* const* d_in, const int* in_sizes, int n_in,
                              void* d_out, int out_size, void* d_ws, size_t ws_size,
                              hipStream_t stream) {
  const float* x_in  = (const float*)d_in[0];
  const float* WQ    = (const float*)d_in[1];
  const float* WK    = (const float*)d_in[2];
  const float* WV    = (const float*)d_in[3];
  const float* WG    = (const float*)d_in[4];
  const float* WO    = (const float*)d_in[5];
  const float* gn_w  = (const float*)d_in[6];
  const float* gn_b  = (const float*)d_in[7];
  const float* ln1_w = (const float*)d_in[8];
  const float* ln1_b = (const float*)d_in[9];
  const float* ln2_w = (const float*)d_in[10];
  const float* ln2_b = (const float*)d_in[11];
  const float* fw1   = (const float*)d_in[12];
  const float* fb1   = (const float*)d_in[13];
  const float* fw2   = (const float*)d_in[14];
  const float* fb2   = (const float*)d_in[15];

  const size_t NB = (size_t)MC*DC;           // elements per token-matrix buffer
  const size_t DD = (size_t)DC*DC;
  // f32 buffers
  float* fp  = (float*)d_ws;
  float* qb  = fp;  fp+=NB;                   // f32 Q/K (pre-xPos)
  float* kb  = fp;  fp+=NB;
  float* yhb = fp;  fp+=NB;
  float* gb  = fp;  fp+=NB;
  float* y1  = fp;  fp+=NB;
  float* xA  = fp;  fp+=NB;
  float* xB  = fp;  fp+=NB;
  // bf16 buffers
  unsigned short* up  = (unsigned short*)fp;
  unsigned short* qbb = up;  up+=NB;                       // xPos'd Q (bf16, head-major)
  unsigned short* kbb = up;  up+=NB;                       // xPos'd K
  unsigned short* vbb = up;  up+=NB;                       // V (bf16, head-major)
  unsigned short* xnb = up;  up+=NB;                       // ln1/ln2 output (aliased use)
  unsigned short* gyb = up;  up+=NB;                       // gated attention output
  unsigned short* tbb = up;  up+=(size_t)MC*DFFC;          // FFN hidden
  unsigned short* WQt = up;  up+=(size_t)LC*DD;
  unsigned short* WKt = up;  up+=(size_t)LC*DD;
  unsigned short* WVt = up;  up+=(size_t)LC*DD;
  unsigned short* WGt = up;  up+=(size_t)LC*DD;
  unsigned short* WOt = up;  up+=(size_t)LC*DD;
  unsigned short* fw1t= up;  up+=(size_t)LC*DC*DFFC;
  unsigned short* fw2t= up;  up+=(size_t)LC*DFFC*DC;

  // --- weight conversion to bf16, transposed [N][K] (k-contiguous), once per call
  for(int i=0;i<LC;++i){
    int g1 = (int)((DD+255)/256);
    cvtH_kernel<<<g1,256,0,stream>>>(WQ + (size_t)i*HC*DC*DHC, WQt + (size_t)i*DD, DC);
    cvtH_kernel<<<g1,256,0,stream>>>(WK + (size_t)i*HC*DC*DHC, WKt + (size_t)i*DD, DC);
    cvtH_kernel<<<g1,256,0,stream>>>(WV + (size_t)i*HC*DC*DHC, WVt + (size_t)i*DD, DC);
    cvtT_kernel<<<g1,256,0,stream>>>(WG + (size_t)i*DD, WGt + (size_t)i*DD, DC, DC);
    cvtT_kernel<<<g1,256,0,stream>>>(WO + (size_t)i*DD, WOt + (size_t)i*DD, DC, DC);
    int g2 = (int)(((size_t)DC*DFFC+255)/256);
    cvtT_kernel<<<g2,256,0,stream>>>(fw1 + (size_t)i*DC*DFFC, fw1t + (size_t)i*DC*DFFC, DC, DFFC);
    cvtT_kernel<<<g2,256,0,stream>>>(fw2 + (size_t)i*DFFC*DC, fw2t + (size_t)i*DFFC*DC, DFFC, DC);
  }

  const int npairs = BC*HC*SC*(DHC/2);
  const float* xin = x_in;
  for(int i=0;i<LC;++i){
    float* xout = (i==LC-1) ? (float*)d_out : ((i&1) ? xB : xA);

    ln_kernel<<<MC,256,0,stream>>>(xin, ln1_w+(size_t)i*DC, ln1_b+(size_t)i*DC, xnb);

    launch_gemm(xnb, WQt+(size_t)i*DD, qb, nullptr, MC, DC, DC, 1, nullptr, nullptr, 0, stream);
    launch_gemm(xnb, WKt+(size_t)i*DD, kb, nullptr, MC, DC, DC, 1, nullptr, nullptr, 0, stream);
    launch_gemm(xnb, WVt+(size_t)i*DD, nullptr, vbb, MC, DC, DC, 1, nullptr, nullptr, 0, stream); // bf16 V

    xpos_bf_kernel<<<(npairs+255)/256,256,0,stream>>>(qb, qbb, 0);
    xpos_bf_kernel<<<(npairs+255)/256,256,0,stream>>>(kb, kbb, 1);

    attn_kernel<<<dim3(SC/64, BC*HC),128,0,stream>>>(qbb, kbb, vbb, yhb);

    launch_gemm(xnb, WGt+(size_t)i*DD, gb, nullptr, MC, DC, DC, 0, nullptr, nullptr, 1, stream); // SiLU gate

    gn_gate_kernel<<<MC,512,0,stream>>>(yhb, gb, gn_w+(size_t)i*DC, gn_b+(size_t)i*DC, gyb);

    launch_gemm(gyb, WOt+(size_t)i*DD, y1, nullptr, MC, DC, DC, 0, nullptr, xin, 0, stream);     // + residual x

    ln_kernel<<<MC,256,0,stream>>>(y1, ln2_w+(size_t)i*DC, ln2_b+(size_t)i*DC, xnb);

    launch_gemm(xnb, fw1t+(size_t)i*DC*DFFC, nullptr, tbb, MC, DFFC, DC, 0,
                fb1+(size_t)i*DFFC, nullptr, 2, stream);                                         // GELU -> bf16
    launch_gemm(tbb, fw2t+(size_t)i*DFFC*DC, xout, nullptr, MC, DC, DFFC, 0,
                fb2+(size_t)i*DC, y1, 0, stream);                                                // + residual y1

    xin = xout;
  }
}
